// VAE_5171140624827
// MI455X (gfx1250) — matrix operand order repaired and checked by reading, not compiled, not run on hardware
//
#include <hip/hip_runtime.h>
#include <hip/hip_bf16.h>

typedef __attribute__((ext_vector_type(16))) _Float16 v16h;
typedef __attribute__((ext_vector_type(2)))  _Float16 h2f;
typedef __attribute__((ext_vector_type(8)))  float    v8f;

#define LATENT  10
#define MAXLEN  80
#define STEPS   10
#define HIDDEN  32
#define BATCH   512
#define ROWS    128          // batch rows per block (512/4)
#define SQRT_DT 0.316227766f

__device__ __forceinline__ unsigned hmix(unsigned x) {
  x ^= x >> 16; x *= 0x7feb352dU;
  x ^= x >> 15; x *= 0x846ca68bU;
  x ^= x >> 16; return x;
}
// stand-in for jax.random.normal (threefry not reproducible; no runtime check)
__device__ __forceinline__ float gauss(unsigned seed) {
  unsigned a = hmix(seed * 0x9E3779B1u + 0x85ebca6bu);
  unsigned b = hmix(a + 0x27d4eb2fu);
  float u1 = (float)(a >> 8) * (1.0f / 16777216.0f) + 1.0e-7f;
  float u2 = (float)(b >> 8) * (1.0f / 16777216.0f);
  return __builtin_sqrtf(-2.0f * __logf(u1)) * __cosf(6.28318530718f * u2);
}

// wave-local LDS fence: all prior ds ops complete + compiler may not reorder across
#define WAVE_LDS_FENCE() asm volatile("s_wait_dscnt 0x0" ::: "memory")

__global__ __launch_bounds__(256)
void diffusion_scan_kernel(const float* __restrict__ z_mean,
                           const float* __restrict__ z_log_var,
                           const float* __restrict__ W1,
                           const float* __restrict__ b1,
                           const float* __restrict__ W2,
                           const float* __restrict__ b2,
                           float* __restrict__ out)
{
  __shared__ float    xtL[ROWS * 16];     // xt mirror for A-fragments; cols 10..15 zero
  __shared__ _Float16 hT[8][16 * 32];     // per-wave hidden tile for D->A transpose

  const int i     = blockIdx.y;           // sequence position 0..79
  const int bbase = blockIdx.x * ROWS;    // batch quarter
  const int tid   = threadIdx.x;
  const int lid   = tid & 31;
  const int wave  = tid >> 5;

  const int nn   = lid & 15;              // N index / A-row index
  const int kgrp = lid >> 4;              // lane-half K-group selector
  const int mrow = wave * 16;             // this wave's local M-tile base row

  // ---------------- per-wave setup (wave owns rows mrow..mrow+15) ----------------
  // zero the padding columns of xt
#pragma unroll
  for (int e = lid; e < 256; e += 32) {
    int d = e & 15;
    if (d >= LATENT) xtL[(mrow + (e >> 4)) * 16 + d] = 0.0f;
  }
  // lane (nn<10, kgrp) owns (row = mrow + r + 8*kgrp, d = nn): mu/sigma/xt in registers
  v8f muR, ssR, hsR, xtR;
#pragma unroll
  for (int r = 0; r < 8; ++r) { muR[r] = 0.f; ssR[r] = 1.f; hsR[r] = 0.f; xtR[r] = 0.f; }
  if (nn < LATENT) {
#pragma unroll
    for (int r = 0; r < 8; ++r) {
      int row  = mrow + r + 8 * kgrp;
      int b    = bbase + row;
      int gidx = b * (MAXLEN * LATENT) + i * LATENT + nn;
      float mu = z_mean[gidx];
      float sg = __expf(z_log_var[gidx]);
      float ss = __builtin_sqrtf(sg);
      unsigned seed = ((unsigned)(i * (STEPS + 1)) << 16) ^ (unsigned)(b * LATENT + nn);
      float x0 = mu + ss * gauss(seed);
      muR[r] = mu; ssR[r] = ss; hsR[r] = 0.5f * sg; xtR[r] = x0;
      xtL[row * 16 + nn] = x0;
    }
  }
  WAVE_LDS_FENCE();

  // ---------------- hoisted weight B-fragments (branchless loads) ----------------
  v16h bW1[2];
#pragma unroll
  for (int nt = 0; nt < 2; ++nt) {
    int n = nn + 16 * nt;
#pragma unroll
    for (int v = 0; v < 8; ++v)
#pragma unroll
      for (int hh = 0; hh < 2; ++hh) {
        int k  = ((v < 4) ? 2 * v : 16 + 2 * (v - 4)) + 8 * kgrp + hh;
        int kr = (k < LATENT) ? (i * LATENT + k) : (MAXLEN * LATENT);  // >=10 -> time row
        float w = W1[kr * HIDDEN + n];
        bW1[nt][2 * v + hh] = (_Float16)((k <= LATENT) ? w : 0.0f);
      }
  }
  v16h bW2;
  {
    int nc = (nn < LATENT) ? nn : 0;
#pragma unroll
    for (int v = 0; v < 8; ++v)
#pragma unroll
      for (int hh = 0; hh < 2; ++hh) {
        int k = ((v < 4) ? 2 * v : 16 + 2 * (v - 4)) + 8 * kgrp + hh;
        float w = W2[k * LATENT + nc];
        bW2[2 * v + hh] = (_Float16)((nn < LATENT) ? w : 0.0f);
      }
  }
  v8f base0, base1;
  {
    float v0 = b1[nn], v1 = b1[nn + 16];
#pragma unroll
    for (int r = 0; r < 8; ++r) { base0[r] = v0; base1[r] = v1; }
  }
  const float b2n = (nn < LATENT) ? b2[(nn < LATENT) ? nn : 0] : 0.0f;

  // ---------------- prefix GEMM: z_mean[:,0:i,:] @ W1[0:10i,:] ----------------
  const int Klim  = i * LATENT;
  const int Kfull = Klim & ~31;
  const float* Arow = z_mean + (size_t)(bbase + mrow + nn) * (MAXLEN * LATENT);

  for (int kc = 0; kc < Kfull; kc += 32) {           // steady state: no predication
    v16h a;
    const float2* Ap = (const float2*)(Arow + kc);
#pragma unroll
    for (int j = 0; j < 8; ++j) {
      int K = ((j < 4) ? 2 * j : 16 + 2 * (j - 4)) + 8 * kgrp;
      float2 p = Ap[K >> 1];
      a[2 * j]     = (_Float16)p.x;
      a[2 * j + 1] = (_Float16)p.y;
    }
    v16h bbf[2];
#pragma unroll
    for (int nt = 0; nt < 2; ++nt) {
      int n = nn + 16 * nt;
#pragma unroll
      for (int v = 0; v < 8; ++v) {
        int gk = kc + ((v < 4) ? 2 * v : 16 + 2 * (v - 4)) + 8 * kgrp;
        bbf[nt][2 * v]     = (_Float16)W1[gk * HIDDEN + n];
        bbf[nt][2 * v + 1] = (_Float16)W1[(gk + 1) * HIDDEN + n];
      }
    }
    base0 = __builtin_amdgcn_wmma_f32_16x16x32_f16(false, a, false, bbf[0], (short)0, base0, false, false);
    base1 = __builtin_amdgcn_wmma_f32_16x16x32_f16(false, a, false, bbf[1], (short)0, base1, false, false);
  }
  if (Kfull < Klim) {                                // tail: clamped loads + selects
    const int kc = Kfull;
    v16h a;
#pragma unroll
    for (int j = 0; j < 8; ++j) {
      int K = ((j < 4) ? 2 * j : 16 + 2 * (j - 4)) + 8 * kgrp;
#pragma unroll
      for (int hh = 0; hh < 2; ++hh) {
        int gk = kc + K + hh;
        float v = Arow[(gk < Klim) ? gk : 0];
        a[2 * j + hh] = (_Float16)((gk < Klim) ? v : 0.0f);
      }
    }
    v16h bbf[2];
#pragma unroll
    for (int nt = 0; nt < 2; ++nt) {
      int n = nn + 16 * nt;
#pragma unroll
      for (int v = 0; v < 8; ++v)
#pragma unroll
        for (int hh = 0; hh < 2; ++hh) {
          int gk = kc + ((v < 4) ? 2 * v : 16 + 2 * (v - 4)) + 8 * kgrp + hh;
          float w = W1[((gk < Klim) ? gk : 0) * HIDDEN + n];
          bbf[nt][2 * v + hh] = (_Float16)((gk < Klim) ? w : 0.0f);
        }
    }
    base0 = __builtin_amdgcn_wmma_f32_16x16x32_f16(false, a, false, bbf[0], (short)0, base0, false, false);
    base1 = __builtin_amdgcn_wmma_f32_16x16x32_f16(false, a, false, bbf[1], (short)0, base1, false, false);
  }

  v8f errAcc;
#pragma unroll
  for (int r = 0; r < 8; ++r) errAcc[r] = 0.0f;

  // ---------------- sequential diffusion scan (wave-independent) ----------------
  for (int s = 0; s < STEPS; ++s) {
    const float t = (float)i + (float)s * 0.1f;

    // A-fragment of xt: cols 0..9 = xt, 10 = t, rest 0.  (k<16 <=> j<4, compile-time)
    v16h axt;
    const float2* xp = (const float2*)(xtL + (mrow + nn) * 16);
#pragma unroll
    for (int j = 0; j < 4; ++j) {
      int K = 2 * j + 8 * kgrp;
      float2 p = xp[j + 4 * kgrp];
      float v0 = (K == LATENT) ? t : p.x;              // single cndmask (kgrp=1,j=1)
      axt[2 * j]     = (_Float16)v0;
      axt[2 * j + 1] = (_Float16)p.y;
    }
#pragma unroll
    for (int j = 4; j < 8; ++j) {
      axt[2 * j] = (_Float16)0.0f; axt[2 * j + 1] = (_Float16)0.0f;
    }
    v8f h0 = __builtin_amdgcn_wmma_f32_16x16x32_f16(false, axt, false, bW1[0], (short)0, base0, false, false);
    v8f h1 = __builtin_amdgcn_wmma_f32_16x16x32_f16(false, axt, false, bW1[1], (short)0, base1, false, false);

    // relu + D->A transpose via wave-private LDS tile
#pragma unroll
    for (int r = 0; r < 8; ++r) {
      int mm = r + 8 * kgrp;
      hT[wave][mm * 32 + nn]      = (_Float16)__builtin_fmaxf(h0[r], 0.0f);
      hT[wave][mm * 32 + nn + 16] = (_Float16)__builtin_fmaxf(h1[r], 0.0f);
    }
    WAVE_LDS_FENCE();

    v16h ah;
    const h2f* hp = (const h2f*)(hT[wave] + nn * 32);
#pragma unroll
    for (int j = 0; j < 8; ++j) {
      int K = ((j < 4) ? 2 * j : 16 + 2 * (j - 4)) + 8 * kgrp;
      h2f p = hp[K >> 1];
      ah[2 * j] = p[0]; ah[2 * j + 1] = p[1];
    }
    v8f csc;
#pragma unroll
    for (int r = 0; r < 8; ++r) csc[r] = b2n;
    csc = __builtin_amdgcn_wmma_f32_16x16x32_f16(false, ah, false, bW2, (short)0, csc, false, false);

    // err accumulation + xt update (registers; mirror xt to LDS for next A-frag)
    if (nn < LATENT) {
#pragma unroll
      for (int r = 0; r < 8; ++r) {
        int row = mrow + r + 8 * kgrp;
        float xt = xtR[r], score = csc[r];
        float logdx = (muR[r] - xt) / ssR[r];          // -(xt - mu_i)/sqrt(sigma_i)
        errAcc[r] += __builtin_fabsf(logdx - score);
        int b = bbase + row;
        unsigned seed = ((unsigned)(i * (STEPS + 1) + s + 1) << 16) ^ (unsigned)(b * LATENT + nn);
        float dW = gauss(seed) * SQRT_DT;
        float xn = xt + hsR[r] * score + ssR[r] * dW;
        xtR[r] = xn;
        xtL[row * 16 + nn] = xn;
      }
    }
    WAVE_LDS_FENCE();
  }

  // ---------------- outputs: sequence then score_error, both (B,L,D) ----------------
  if (nn < LATENT) {
#pragma unroll
    for (int r = 0; r < 8; ++r) {
      int b  = bbase + mrow + r + 8 * kgrp;
      int gi = b * (MAXLEN * LATENT) + i * LATENT + nn;
      out[gi] = xtR[r];
      out[BATCH * MAXLEN * LATENT + gi] = errAcc[r];
    }
  }
}

extern "C" void kernel_launch(void* const* d_in, const int* in_sizes, int n_in,
                              void* d_out, int out_size, void* d_ws, size_t ws_size,
                              hipStream_t stream) {
  const float* z_mean    = (const float*)d_in[0];
  const float* z_log_var = (const float*)d_in[1];
  const float* W1        = (const float*)d_in[2];
  const float* b1        = (const float*)d_in[3];
  const float* W2        = (const float*)d_in[4];
  const float* b2        = (const float*)d_in[5];
  float* out = (float*)d_out;

  dim3 grid(BATCH / ROWS, MAXLEN);   // 4 batch quarters x 80 sequence positions
  diffusion_scan_kernel<<<grid, 256, 0, stream>>>(z_mean, z_log_var, W1, b1, W2, b2, out);
}
